// GBGraphConvModel_8074538516964
// MI455X (gfx1250) — matrix operand order repaired and checked by reading, not compiled, not run on hardware
//
#include <hip/hip_runtime.h>
#include <hip/hip_bf16.h>

#define N_ATOMS 122880
#define BATCH   4096
#define EPS_BN  1e-3f

typedef __attribute__((ext_vector_type(16))) __bf16 v16bf;
typedef __attribute__((ext_vector_type(8)))  float  v8f;
typedef __attribute__((ext_vector_type(4)))  unsigned int u32x4;
typedef __attribute__((ext_vector_type(8)))  int i32x8;
typedef __attribute__((ext_vector_type(4)))  int i32x4;

#if defined(__has_builtin)
#  if __has_builtin(__builtin_amdgcn_tensor_load_to_lds)
#    define HAVE_TDM 1
#  endif
#endif

__constant__ int c_offs[12] = {0,1024,25600,62464,99328,119808,121856,122368,122624,122752,122816,122880};

struct Adjs { const int* a[10]; };

__device__ __forceinline__ int deg_of(int i){
  int d = 0;
  #pragma unroll
  for (int k = 1; k <= 10; ++k) d += (i >= c_offs[k]) ? 1 : 0;
  return d;
}

__device__ __forceinline__ unsigned short f2bf(float f){
  unsigned int u = __float_as_uint(f);
  u += 0x7FFFu + ((u >> 16) & 1u);           // round-to-nearest-even
  return (unsigned short)(u >> 16);
}

// order-preserving float <-> uint key for atomic max
__device__ __forceinline__ unsigned int fkey(float x){
  unsigned int u = __float_as_uint(x);
  return (u & 0x80000000u) ? ~u : (u | 0x80000000u);
}
__device__ __forceinline__ float fdecode(unsigned int k){
  unsigned int u = (k & 0x80000000u) ? (k & 0x7FFFFFFFu) : ~k;
  return __uint_as_float(u);
}

union ABu { v16bf v; unsigned int u[8]; unsigned short s[16]; };

// ---------------------------------------------------------------------------
// Weight pre-swizzle: per-degree stacked [W_rel ; W_self] in a contiguous-k 16-bit
// B-matrix (32x16) fragment layout: [deg][ntile(4)][kstep][lane][16]
// k_in_tile = (lane>>4)*16 + j ; n = ntile*16 + (lane&15)
// ---------------------------------------------------------------------------
__global__ void prep_gc_weights(const float* __restrict__ W, const float* __restrict__ b,
                                int F, int KH, int KS,
                                unsigned short* __restrict__ Bz, float* __restrict__ bias){
  long idx = (long)blockIdx.x * blockDim.x + threadIdx.x;
  long total = 11L * 4 * KS * 512;
  if (idx < total){
    int j    = (int)(idx & 15);
    int lane = (int)((idx >> 4) & 31);
    long rest = idx >> 9;                  // fragment index
    int kk  = (int)(rest % KS);
    int t   = (int)((rest / KS) & 3);
    int deg = (int)(rest / (KS * 4));
    int half = lane >> 4, N = lane & 15;
    int k = kk * 32 + half * 16 + j;
    int n = t * 16 + N;
    float w = 0.f;
    if (k < KH){                            // rel block
      if (deg > 0 && k < F) w = W[((size_t)(2*(deg-1))*F + k)*64 + n];
    } else {                                // self block
      int f = k - KH;
      if (f < F){
        int wi = (deg > 0) ? (2*deg - 1) : 20;
        w = W[((size_t)wi*F + f)*64 + n];
      }
    }
    Bz[idx] = f2bf(w);
  }
  if (idx < 11 * 64){
    int deg = (int)(idx / 64), n = (int)(idx % 64);
    bias[idx] = (deg > 0) ? (b[(2*(deg-1))*64 + n] + b[(2*deg-1)*64 + n]) : b[20*64 + n];
  }
}

// d1_W (64x128) -> fragment layout [ntile(8)][kstep(2)][lane][16]
__global__ void prep_d1_weights(const float* __restrict__ W, unsigned short* __restrict__ Bz){
  int idx = blockIdx.x * blockDim.x + threadIdx.x;
  if (idx >= 8 * 2 * 512) return;
  int j    = idx & 15;
  int lane = (idx >> 4) & 31;
  int rest = idx >> 9;
  int kk = rest & 1;
  int t  = rest >> 1;
  int k = kk * 32 + (lane >> 4) * 16 + j;
  int n = t * 16 + (lane & 15);
  Bz[idx] = f2bf(W[(size_t)k * 128 + n]);
}

// ---------------------------------------------------------------------------
// Build bf16 concat activations: out[i, 0:KH] = sum_j src[adj[i][j]]  (rel)
//                                out[i, KH:2KH] = src[i]              (self)
// ---------------------------------------------------------------------------
__global__ void build_cat_kernel(const float* __restrict__ src, int F, int KH, Adjs adjs,
                                 unsigned short* __restrict__ out){
  long idx = (long)blockIdx.x * blockDim.x + threadIdx.x;
  int Kpad = 2 * KH;
  if (idx >= (long)N_ATOMS * Kpad) return;
  int i = (int)(idx / Kpad);
  int c = (int)(idx % Kpad);
  int d = deg_of(i);
  float v = 0.f;
  if (c < KH){
    if (d > 0 && c < F){
      int li = i - c_offs[d];
      const int* nb = adjs.a[d-1] + (long)li * d;
      for (int j = 0; j < d; ++j) v += src[(long)nb[j] * F + c];
    }
  } else {
    int f = c - KH;
    if (f < F) v = src[(long)i * F + f];
  }
  out[idx] = f2bf(v);
}

// ---------------------------------------------------------------------------
// WMMA GEMM + bias + tanh + batchnorm.
// Block = 128 threads = 4 waves = 64 rows; every segment offset is a multiple
// of 64, so each block is degree-uniform.  The block's B-fragment set is DMA'd
// into LDS once (TDM tensor_load_to_lds when available), then all 4 waves run
// the WMMA K-loop out of LDS.
// ---------------------------------------------------------------------------
template<int NT, int KS, bool PER_DEG>
__global__ void gemm_tanh_bn_kernel(const unsigned short* __restrict__ A,
                                    const unsigned short* __restrict__ Bz,
                                    const float* __restrict__ bias,
                                    const float* __restrict__ g,  const float* __restrict__ be,
                                    const float* __restrict__ mu, const float* __restrict__ var,
                                    float* __restrict__ out, int ncols){
  __shared__ __align__(64) unsigned short sB[NT * KS * 512];

  const int lane = threadIdx.x & 31;
  const int wave = threadIdx.x >> 5;
  const int tile = blockIdx.x * (blockDim.x >> 5) + wave;
  const int row0 = tile * 16;
  if (row0 >= N_ATOMS) return;
  const int Kpad = KS * 32;
  const int deg  = PER_DEG ? deg_of(blockIdx.x * 64) : 0;   // uniform per block
  const unsigned short* gB = Bz + (size_t)deg * NT * KS * 512;

#ifdef HAVE_TDM
  if (wave == 0){
    // 1-D TDM copy: NT*KS*1024 bytes of B fragments, data_size = 8B units
    const unsigned int units = (unsigned int)(NT * KS * 512 * 2 / 8);
    unsigned long long ga = (unsigned long long)(uintptr_t)gB;
    u32x4 g0 = { 1u,                                   // count=1 valid user D#
                 (unsigned int)(uintptr_t)sB,          // lds_addr (bytes)
                 (unsigned int)ga,                     // global_addr[31:0]
                 (unsigned int)((ga >> 32) & 0x01FFFFFFu) | (2u << 30) }; // addr[56:32], type=2
    i32x8 g1 = { (int)(3u << 16),                      // data_size = 8 bytes
                 (int)((units & 0xFFFFu) << 16),       // tensor_dim0[15:0]
                 (int)(units >> 16),                   // tensor_dim0[31:16]
                 (int)((units & 0xFFFFu) << 16),       // tile_dim0
                 0,                                    // tile_dim1=0 (1-D), tile_dim2=0
                 (int)units,                           // tensor_dim0_stride lo
                 0, 0 };
    i32x4 z4 = { 0, 0, 0, 0 };
    i32x8 z8 = { 0, 0, 0, 0, 0, 0, 0, 0 };
    __builtin_amdgcn_tensor_load_to_lds(g0, g1, z4, z4, z8, 0);
#if __has_builtin(__builtin_amdgcn_s_wait_tensorcnt)
    __builtin_amdgcn_s_wait_tensorcnt(0);
#else
    asm volatile("s_wait_tensorcnt 0x0" ::: "memory");
#endif
  }
  __syncthreads();
#else
  for (int i = threadIdx.x; i < NT * KS * 64; i += blockDim.x)
    ((uint4*)sB)[i] = ((const uint4*)gB)[i];
  __syncthreads();
#endif

  const int M = lane & 15, half = lane >> 4;
  const unsigned short* arow = A + (size_t)(row0 + M) * Kpad;

  v8f acc[NT];
  #pragma unroll
  for (int t = 0; t < NT; ++t)
    #pragma unroll
    for (int e = 0; e < 8; ++e) acc[t][e] = 0.f;

  #pragma unroll
  for (int kk = 0; kk < KS; ++kk){
    // A fragment (contiguous k, same order as B): VGPR v holds K = half*16+v*2 (+1)
    ABu a;
    #pragma unroll
    for (int j = 0; j < 8; ++j){
      int k = kk*32 + (half << 4) + (j << 1);
      a.u[j] = *(const unsigned int*)(arow + k);
    }
    #pragma unroll
    for (int t = 0; t < NT; ++t){
      const v16bf* bp = (const v16bf*)(sB + ((t*KS + kk)*32 + lane)*16);
      acc[t] = __builtin_amdgcn_wmma_f32_16x16x32_bf16(false, a.v, false, *bp,
                                                       (short)0, acc[t], false, false);
    }
  }

  // C/D layout: lanes 0-15 -> M=v, lanes 16-31 -> M=v+8 ; N = (lane&15)+t*16
  #pragma unroll
  for (int t = 0; t < NT; ++t){
    int n = t*16 + M;
    float bs = bias[(PER_DEG ? deg*ncols : 0) + n];
    float gg = g[n], bb = be[n], mm = mu[n];
    float iv = rsqrtf(var[n] + EPS_BN);
    #pragma unroll
    for (int v = 0; v < 8; ++v){
      int m = v + half*8;
      float x = tanhf(acc[t][v] + bs);
      out[(size_t)(row0 + m)*ncols + n] = gg * (x - mm) * iv + bb;
    }
  }
}

// pool: p[i] = max(h[i], max_j h[adj[i][j]])  (64 features)
__global__ void pool_kernel(const float* __restrict__ h, Adjs adjs, float* __restrict__ p){
  long idx = (long)blockIdx.x * blockDim.x + threadIdx.x;
  if (idx >= (long)N_ATOMS * 64) return;
  int i = (int)(idx >> 6), f = (int)(idx & 63);
  int d = deg_of(i);
  float v = h[idx];
  if (d > 0){
    int li = i - c_offs[d];
    const int* nb = adjs.a[d-1] + (long)li * d;
    for (int j = 0; j < d; ++j) v = fmaxf(v, h[((long)nb[j] << 6) + f]);
  }
  p[idx] = v;
}

__global__ void f32_to_bf16_kernel(const float* __restrict__ s, unsigned short* __restrict__ d, long n){
  long i = (long)blockIdx.x * blockDim.x + threadIdx.x;
  if (i < n) d[i] = f2bf(s[i]);
}

__global__ void init_seg_kernel(float* __restrict__ ssum, unsigned int* __restrict__ smax){
  int i = blockIdx.x * blockDim.x + threadIdx.x;
  if (i >= BATCH * 128) return;
  ssum[i] = 0.f;
  smax[i] = fkey(-__builtin_inff());
}

__global__ void segment_kernel(const float* __restrict__ hd, const int* __restrict__ mem,
                               float* __restrict__ ssum, unsigned int* __restrict__ smax){
  long idx = (long)blockIdx.x * blockDim.x + threadIdx.x;
  if (idx >= (long)N_ATOMS * 128) return;
  int i = (int)(idx >> 7), f = (int)(idx & 127);
  float x = hd[idx];
  int m = mem[i];
  atomicAdd(&ssum[(size_t)m*128 + f], x);
  atomicMax(&smax[(size_t)m*128 + f], fkey(x));
}

__global__ void final_kernel(const float* __restrict__ ssum, const unsigned int* __restrict__ smax,
                             const float* __restrict__ d2W, const float* __restrict__ d2b,
                             const float* __restrict__ d3W, const float* __restrict__ d3b,
                             const float* __restrict__ xadd, float* __restrict__ out){
  int t = blockIdx.x * blockDim.x + threadIdx.x;
  if (t >= BATCH) return;
  float acc = d2b[0];
  for (int f = 0; f < 128; ++f) acc += tanhf(ssum[(size_t)t*128 + f]) * d2W[f];
  for (int f = 0; f < 128; ++f) acc += tanhf(fdecode(smax[(size_t)t*128 + f])) * d2W[128 + f];
  float r = d3b[0] + acc * d3W[0];
  for (int j = 0; j < 15; ++j) r += xadd[(size_t)t*15 + j] * d3W[j + 1];
  out[t] = r;
}

static inline int cdiv(long a, long b){ return (int)((a + b - 1) / b); }

extern "C" void kernel_launch(void* const* d_in, const int* in_sizes, int n_in,
                              void* d_out, int out_size, void* d_ws, size_t ws_size,
                              hipStream_t stream){
  (void)in_sizes; (void)n_in; (void)out_size; (void)ws_size;
  const float* atoms = (const float*)d_in[0];
  const int* membership = (const int*)d_in[1];
  Adjs adjs;
  for (int d = 0; d < 10; ++d) adjs.a[d] = (const int*)d_in[2 + d];
  const float* x_add = (const float*)d_in[12];
  const float* gc1_W = (const float*)d_in[13];
  const float* gc1_b = (const float*)d_in[14];
  const float* gc2_W = (const float*)d_in[15];
  const float* gc2_b = (const float*)d_in[16];
  const float* bn1_g = (const float*)d_in[17];
  const float* bn1_be= (const float*)d_in[18];
  const float* bn1_m = (const float*)d_in[19];
  const float* bn1_v = (const float*)d_in[20];
  const float* bn3_g = (const float*)d_in[21];
  const float* bn3_be= (const float*)d_in[22];
  const float* bn3_m = (const float*)d_in[23];
  const float* bn3_v = (const float*)d_in[24];
  const float* d1_W = (const float*)d_in[25];
  const float* d1_b = (const float*)d_in[26];
  const float* d2_W = (const float*)d_in[27];
  const float* d2_b = (const float*)d_in[28];
  const float* d3_W = (const float*)d_in[29];
  const float* d3_b = (const float*)d_in[30];
  float* out = (float*)d_out;

  char* base = (char*)d_ws; size_t off = 0;
  auto carve = [&](size_t bytes) -> char* {
    char* p = base + off; off += (bytes + 255) & ~(size_t)255; return p;
  };
  unsigned short* Bz1  = (unsigned short*)carve(11ul*4*5*512*2);   // gc1 weight frags
  unsigned short* Bz2  = (unsigned short*)carve(11ul*4*4*512*2);   // gc2 weight frags
  unsigned short* BzD  = (unsigned short*)carve(8ul*2*512*2);      // d1 weight frags
  float* bias1 = (float*)carve(11ul*64*4);
  float* bias2 = (float*)carve(11ul*64*4);
  unsigned short* Acat = (unsigned short*)carve((size_t)N_ATOMS*160*2); // concat bf16 (reused)
  float* H  = (float*)carve((size_t)N_ATOMS*64*4);
  float* P  = (float*)carve((size_t)N_ATOMS*64*4);
  float* HD = (float*)carve((size_t)N_ATOMS*128*4);
  float* SSUM = (float*)carve((size_t)BATCH*128*4);
  unsigned int* SMAX = (unsigned int*)carve((size_t)BATCH*128*4);

  const int TB = 256;
  const int GEMM_TB = 128;                 // 4 waves, 64 rows -> degree-uniform
  const int GEMM_BLOCKS = N_ATOMS / 64;    // 1920

  // weight prep
  prep_gc_weights<<<cdiv(11L*4*5*512, TB), TB, 0, stream>>>(gc1_W, gc1_b, 75, 80, 5, Bz1, bias1);
  prep_gc_weights<<<cdiv(11L*4*4*512, TB), TB, 0, stream>>>(gc2_W, gc2_b, 64, 64, 4, Bz2, bias2);
  prep_d1_weights<<<cdiv(8L*2*512, TB), TB, 0, stream>>>(d1_W, BzD);

  // gc1: concat -> WMMA gemm (+tanh+bn1) -> pool
  build_cat_kernel<<<cdiv((long)N_ATOMS*160, TB), TB, 0, stream>>>(atoms, 75, 80, adjs, Acat);
  gemm_tanh_bn_kernel<4,5,true><<<GEMM_BLOCKS, GEMM_TB, 0, stream>>>(Acat, Bz1, bias1,
      bn1_g, bn1_be, bn1_m, bn1_v, H, 64);
  pool_kernel<<<cdiv((long)N_ATOMS*64, TB), TB, 0, stream>>>(H, adjs, P);

  // gc2
  build_cat_kernel<<<cdiv((long)N_ATOMS*128, TB), TB, 0, stream>>>(P, 64, 64, adjs, Acat);
  gemm_tanh_bn_kernel<4,4,true><<<GEMM_BLOCKS, GEMM_TB, 0, stream>>>(Acat, Bz2, bias2,
      bn1_g, bn1_be, bn1_m, bn1_v, H, 64);
  pool_kernel<<<cdiv((long)N_ATOMS*64, TB), TB, 0, stream>>>(H, adjs, P);

  // d1: 64 -> 128 (+tanh+bn3)
  f32_to_bf16_kernel<<<cdiv((long)N_ATOMS*64, TB), TB, 0, stream>>>(P, Acat, (long)N_ATOMS*64);
  gemm_tanh_bn_kernel<8,2,false><<<GEMM_BLOCKS, GEMM_TB, 0, stream>>>(Acat, BzD, d1_b,
      bn3_g, bn3_be, bn3_m, bn3_v, HD, 128);

  // molecule segments + head
  init_seg_kernel<<<cdiv((long)BATCH*128, TB), TB, 0, stream>>>(SSUM, SMAX);
  segment_kernel<<<cdiv((long)N_ATOMS*128, TB), TB, 0, stream>>>(HD, membership, SSUM, SMAX);
  final_kernel<<<cdiv((long)BATCH, TB), TB, 0, stream>>>(SSUM, SMAX, d2_W, d2_b, d3_W, d3_b, x_add, out);
}
